// SACRSN_v43_23536420782610
// MI455X (gfx1250) — compile-verified, matrix operand order checked by reading, not run on hardware
//
#include <hip/hip_runtime.h>

#define B_  128
#define T_  32
#define D_  64
#define S_  32
#define V_  8192
#define TWOD 128
#define TWOPI 6.283185307179586f
#define MV ((size_t)4096 * 8192)

typedef __attribute__((ext_vector_type(16))) __bf16 v16bf;
typedef __attribute__((ext_vector_type(8)))  float  v8f;
typedef __attribute__((ext_vector_type(4)))  int    v4i;

#define AS1 __attribute__((address_space(1)))
#define AS3 __attribute__((address_space(3)))

#if __has_builtin(__builtin_amdgcn_global_load_async_to_lds_b128) && \
    __has_builtin(__builtin_amdgcn_s_wait_asynccnt)
#define HAS_ASYNC_LDS 1
#else
#define HAS_ASYNC_LDS 0
#endif

struct P {
  // inputs (setup_inputs order)
  const int*   x_seq;   const float* enc;     const float* vq;
  const float* nrg;     const float* nrb;     const float* nig;  const float* nib;
  const float* qkv_Wr;  const float* qkv_br;  const float* qkv_Wi; const float* qkv_bi;
  const float* gate_w;  const float* gate_b;  const float* arb_W;  const float* arb_b;
  const float* dec_W;   const float* dec_b;   const float* halt_bias; const float* input_gate;
  // workspace
  float *x_r, *x_i, *gw_r, *gw_i, *mem_r, *mem_i, *z_f32, *prev_ang;
  float *active, *acc_vq, *acc_ent, *acc_pond, *acc_ph;
  int   *fin_idx;
  float *vql, *sim, *m_r, *m_i, *qkv, *e_sq, *part_val;
  int   *part_idx, *idx, *counts;
  float *zq, *stats, *b_re, *b_im;
  __bf16 *z_bf, *vq_bf, *dec_bf, *out_bf, *Wre, *Wim;
  float *out; // d_out
};

// ---- WMMA helpers (CDNA5 bf16 16x16x32, wave32) -----------------------------
// Operand layout (ISA 7.12.2, 16-bit A 16x32): lane l -> row (l&15),
// k elements { koff..koff+7 , koff+16..koff+23 } with koff = (l>>4)*8.
// Transposed-B (N x K row-major) loads identically with n = l&15.
__device__ __forceinline__ v16bf ld_op(const __bf16* p) {
  v16bf v;
#pragma unroll
  for (int i = 0; i < 8; ++i) { v[i] = p[i]; v[i + 8] = p[16 + i]; }
  return v;
}
__device__ __forceinline__ v8f wmma_bf16(v16bf a, v16bf b, v8f c) {
  return __builtin_amdgcn_wmma_f32_16x16x32_bf16(false, a, false, b, (short)0, c, false, false);
}
__device__ __forceinline__ float sigf(float x) { return 1.f / (1.f + expf(-x)); }

// Cooperative global->LDS stage: `chunks16` 16-byte chunks, 256 threads.
// Async path issues GLOBAL_LOAD_ASYNC_TO_LDS_B128 then waits for ASYNCcnt == 0 before
// returning; caller must __syncthreads() after this returns.
__device__ __forceinline__ void stage_to_lds(const char* g, char* l, int chunks16, int tid) {
#if HAS_ASYNC_LDS
  for (int i = tid; i < chunks16; i += 256) {
    __builtin_amdgcn_global_load_async_to_lds_b128(
        (AS1 v4i*)(g + (size_t)i * 16), (AS3 v4i*)(l + (size_t)i * 16), 0, 0);
  }
  __builtin_amdgcn_s_wait_asynccnt(0);
#else
  const float4* s4 = (const float4*)g;
  float4* d4 = (float4*)l;
  for (int i = tid; i < chunks16; i += 256) d4[i] = s4[i];
#endif
}

// ---- prep kernels -----------------------------------------------------------
__global__ void prep_zero(P p) {
  int stride = gridDim.x * blockDim.x;
  for (int i = blockIdx.x * blockDim.x + threadIdx.x; i < B_ * S_ * D_; i += stride) {
    if (i < B_ * D_) { p.gw_r[i] = 0.f; p.gw_i[i] = 0.f; }
    p.mem_r[i] = 0.f; p.mem_i[i] = 0.f;
  }
  if (blockIdx.x == 0 && threadIdx.x < 4) p.stats[threadIdx.x] = 0.f;
}

__global__ void prep_cvt(P p) {
  int stride = gridDim.x * blockDim.x;
  for (int i = blockIdx.x * blockDim.x + threadIdx.x; i < V_ * TWOD; i += stride) {
    p.vq_bf[i]  = (__bf16)p.vq[i];
    p.dec_bf[i] = (__bf16)p.dec_W[i];
  }
}

__global__ void prep_esq(P p) {
  int n = blockIdx.x * blockDim.x + threadIdx.x;
  if (n < V_) {
    float s = 0.f;
    for (int k = 0; k < TWOD; ++k) { float v = p.vq[(size_t)n * TWOD + k]; s += v * v; }
    p.e_sq[n] = s;
  }
}

// pack complex weights: real = [Wr | -Wi], imag = [Wi | Wr]; biases br-bi / br+bi
__global__ void prep_packw(P p) {
  int g = blockIdx.x * blockDim.x + threadIdx.x;
  if (g < 3 * D_ * TWOD) {
    int h = g / (D_ * TWOD), rem = g % (D_ * TWOD);
    int n = rem / TWOD, k = rem % TWOD;
    const float* Wr = p.qkv_Wr + (size_t)h * D_ * D_ + n * D_;
    const float* Wi = p.qkv_Wi + (size_t)h * D_ * D_ + n * D_;
    p.Wre[g] = (__bf16)((k < D_) ? Wr[k] : -Wi[k - D_]);
    p.Wim[g] = (__bf16)((k < D_) ? Wi[k] :  Wr[k - D_]);
  }
  if (g < 3 * D_) {
    p.b_re[g] = p.qkv_br[g] - p.qkv_bi[g];
    p.b_im[g] = p.qkv_br[g] + p.qkv_bi[g];
  }
}

__global__ void embed_k(P p) {
  int g = blockIdx.x * blockDim.x + threadIdx.x;
  if (g >= B_ * T_) return;
  int b = g >> 5, t = g & 31;
  int tok = p.x_seq[g];
  const float* e = p.enc + (size_t)tok * TWOD;
  float* xr = p.x_r + ((size_t)t * B_ + b) * D_;
  float* xi = p.x_i + ((size_t)t * B_ + b) * D_;
  for (int d = 0; d < D_; ++d) { xr[d] = e[d]; xi[d] = e[D_ + d]; }
}

// ---- per-(t,rec) stage 1: blend + LayerNorm + bf16 pack ---------------------
__global__ void step_pre(P p, int t, int rec) {
  int b = threadIdx.x;
  if (b >= B_) return;
  float* gr = p.gw_r + b * D_;
  float* gi = p.gw_i + b * D_;
  if (rec == 0) {
    float alpha = sigf(p.input_gate[0]);
    const float* xr = p.x_r + ((size_t)t * B_ + b) * D_;
    const float* xi = p.x_i + ((size_t)t * B_ + b) * D_;
    for (int d = 0; d < D_; ++d) {
      gr[d] = alpha * gr[d] + (1.f - alpha) * xr[d];
      gi[d] = alpha * gi[d] + (1.f - alpha) * xi[d];
      p.prev_ang[b * D_ + d] = atan2f(gi[d], gr[d]);
    }
    p.active[b] = 1.f; p.acc_vq[b] = 0.f; p.acc_ent[b] = 0.f;
    p.acc_pond[b] = 0.f; p.acc_ph[b] = 0.f; p.fin_idx[b] = 0;
  }
  // LayerNorm real
  float mu = 0.f; for (int d = 0; d < D_; ++d) mu += gr[d];
  mu *= (1.f / D_);
  float var = 0.f; for (int d = 0; d < D_; ++d) { float c = gr[d] - mu; var += c * c; }
  var *= (1.f / D_);
  float rs = rsqrtf(var + 1e-5f);
  for (int d = 0; d < D_; ++d) {
    float c = (gr[d] - mu) * rs * p.nrg[d] + p.nrb[d];
    p.z_f32[b * TWOD + d] = c; p.z_bf[b * TWOD + d] = (__bf16)c;
  }
  // LayerNorm imag
  mu = 0.f; for (int d = 0; d < D_; ++d) mu += gi[d];
  mu *= (1.f / D_);
  var = 0.f; for (int d = 0; d < D_; ++d) { float c = gi[d] - mu; var += c * c; }
  var *= (1.f / D_);
  rs = rsqrtf(var + 1e-5f);
  for (int d = 0; d < D_; ++d) {
    float c = (gi[d] - mu) * rs * p.nig[d] + p.nib[d];
    p.z_f32[b * TWOD + D_ + d] = c; p.z_bf[b * TWOD + D_ + d] = (__bf16)c;
  }
}

// ---- per-(t,rec) stage 2: VQ search, 64 WGs over V --------------------------
// score = z.e - 0.5*||e||^2 (argmax == argmin d2). Per-WG: 128 codes, staged
// once into LDS via async copy (B tiles are shared by all 8 waves).
__global__ void vq_search(P p) {
  __shared__ __align__(16) __bf16 Bt[128 * TWOD]; // 32 KB code block
  int tid = threadIdx.x;
  int lane = tid & 31, w = tid >> 5;
  int nbase = blockIdx.x * 128;
  int r = lane & 15, half = lane >> 4;

  stage_to_lds((const char*)(p.vq_bf + (size_t)nbase * TWOD), (char*)Bt,
               (128 * TWOD * 2) / 16, tid);

  // A operands (per-wave M strip) load from global after the stage completes
  const __bf16* arow = p.z_bf + ((size_t)(16 * w + r)) * TWOD + half * 8;
  v16bf a[4];
#pragma unroll
  for (int kk = 0; kk < 4; ++kk) a[kk] = ld_op(arow + 32 * kk);
  __syncthreads();

  float bestv[8]; int besti[8];
#pragma unroll
  for (int v = 0; v < 8; ++v) { bestv[v] = -3.4e38f; besti[v] = 0; }
  for (int nt = 0; nt < 8; ++nt) {
    int n = nbase + nt * 16 + r;
    const __bf16* brow = Bt + (size_t)(nt * 16 + r) * TWOD + half * 8;
    v8f acc = {0.f, 0.f, 0.f, 0.f, 0.f, 0.f, 0.f, 0.f};
#pragma unroll
    for (int kk = 0; kk < 4; ++kk) acc = wmma_bf16(a[kk], ld_op(brow + 32 * kk), acc);
    float sh = -0.5f * p.e_sq[n];
#pragma unroll
    for (int v = 0; v < 8; ++v) {
      float s = acc[v] + sh;
      if (s > bestv[v]) { bestv[v] = s; besti[v] = n; }
    }
  }
  // reduce over the 16 lanes of each half (each half owns distinct rows)
#pragma unroll
  for (int v = 0; v < 8; ++v) {
    float bv = bestv[v]; int bi = besti[v];
    for (int off = 1; off < 16; off <<= 1) {
      float ov = __shfl_xor(bv, off);
      int   oi = __shfl_xor(bi, off);
      if (ov > bv || (ov == bv && oi < bi)) { bv = ov; bi = oi; }
    }
    if (r == 0) {
      int m = 16 * w + v + half * 8;
      p.part_val[blockIdx.x * B_ + m] = bv;
      p.part_idx[blockIdx.x * B_ + m] = bi;
    }
  }
}

// ---- per-(t,rec) stage 3: everything else (1 WG, 8 waves) -------------------
__global__ void step_main(P p, int t, int rec) {
  __shared__ float s_red[B_];
  __shared__ float s_ent;
  int tid = threadIdx.x;
  int lane = tid & 31, w = tid >> 5;

  // A: zero histogram; reduce VQ partials
  for (int j = tid; j < V_; j += 256) p.counts[j] = 0;
  if (tid < B_) {
    float bv = -3.4e38f; int bi = 0;
    for (int blk = 0; blk < 64; ++blk) {
      float v = p.part_val[blk * B_ + tid];
      if (v > bv) { bv = v; bi = p.part_idx[blk * B_ + tid]; }
    }
    p.idx[tid] = bi;
  }
  __syncthreads();
  if (tid < B_) atomicAdd(&p.counts[p.idx[tid]], 1);
  __syncthreads();
  // C: z_q gather, vq_loss, entropy terms
  if (tid < B_) {
    int ii = p.idx[tid];
    float ss = 0.f;
    for (int k = 0; k < TWOD; ++k) {
      float q = p.vq[(size_t)ii * TWOD + k];
      p.zq[tid * TWOD + k] = q;
      float d = q - p.z_f32[tid * TWOD + k];
      ss += d * d;
    }
    p.vql[tid] = 1.25f * ss * (1.f / TWOD);
    s_red[tid] = logf((float)p.counts[ii] * (1.f / B_) + 1e-10f);
  }
  __syncthreads();
  if (tid == 0) {
    float s = 0.f;
    for (int i = 0; i < B_; ++i) s += s_red[i];
    s_ent = -s * (1.f / B_);
  }
  __syncthreads();

  // D: QKV via WMMA — A = [cr|ci] bf16, B = packed complex weights (K=128)
  {
    int r = lane & 15, half = lane >> 4;
    const __bf16* arow = p.z_bf + ((size_t)(16 * w + r)) * TWOD + half * 8;
    v16bf a[4];
#pragma unroll
    for (int kk = 0; kk < 4; ++kk) a[kk] = ld_op(arow + 32 * kk);
    for (int o = 0; o < 6; ++o) {
      int h = o >> 1;
      const __bf16* Wp   = ((o & 1) ? p.Wim : p.Wre) + (size_t)h * D_ * TWOD;
      const float*  bias = ((o & 1) ? p.b_im : p.b_re) + h * D_;
      float* outp = p.qkv + (size_t)o * B_ * D_;
      for (int nt = 0; nt < 4; ++nt) {
        const __bf16* brow = Wp + ((size_t)(nt * 16 + r)) * TWOD + half * 8;
        v8f acc = {0.f, 0.f, 0.f, 0.f, 0.f, 0.f, 0.f, 0.f};
#pragma unroll
        for (int kk = 0; kk < 4; ++kk) acc = wmma_bf16(a[kk], ld_op(brow + 32 * kk), acc);
        int n = nt * 16 + r;
#pragma unroll
        for (int v = 0; v < 8; ++v) {
          int m = 16 * w + v + half * 8;
          outp[m * D_ + n] = acc[v] + bias[n];
        }
      }
    }
  }
  __syncthreads();

  // F1: attention scores sim[b][s]
  for (int c = tid; c < B_ * S_; c += 256) {
    int b = c >> 5, s = c & 31;
    const float* mr = p.mem_r + (size_t)b * S_ * D_ + s * D_;
    const float* mi = p.mem_i + (size_t)b * S_ * D_ + s * D_;
    float acc = 0.f;
    for (int d = 0; d < D_; ++d)
      acc += mr[d] * p.z_f32[b * TWOD + d] + mi[d] * p.z_f32[b * TWOD + D_ + d];
    p.sim[c] = acc;
  }
  __syncthreads();
  // F2: softmax over S
  if (tid < B_) {
    float mx = -3.4e38f;
    for (int s = 0; s < S_; ++s) mx = fmaxf(mx, p.sim[tid * S_ + s]);
    float sum = 0.f;
    for (int s = 0; s < S_; ++s) { float e = expf(p.sim[tid * S_ + s] - mx); p.sim[tid * S_ + s] = e; sum += e; }
    float inv = 1.f / sum;
    for (int s = 0; s < S_; ++s) p.sim[tid * S_ + s] *= inv;
  }
  __syncthreads();
  // F3: attended memory
  for (int c = tid; c < B_ * D_; c += 256) {
    int b = c >> 6, d = c & 63;
    float sr = 0.f, si = 0.f;
    for (int s = 0; s < S_; ++s) {
      float a = p.sim[b * S_ + s];
      sr += a * p.mem_r[(size_t)b * S_ * D_ + s * D_ + d];
      si += a * p.mem_i[(size_t)b * S_ * D_ + s * D_ + d];
    }
    p.m_r[c] = sr; p.m_i[c] = si;
  }
  __syncthreads();

  // H: gates, update, phase, halting (one thread per row)
  if (tid < B_) {
    int b = tid;
    const float* qr = p.qkv + 0 * B_ * D_ + b * D_;
    const float* qi = p.qkv + 1 * B_ * D_ + b * D_;
    const float* kr = p.qkv + 2 * B_ * D_ + b * D_;
    const float* ki = p.qkv + 3 * B_ * D_ + b * D_;
    const float* vr = p.qkv + 4 * B_ * D_ + b * D_;
    const float* vi = p.qkv + 5 * B_ * D_ + b * D_;
    float gdot = 0.f;
    for (int d = 0; d < D_; ++d) gdot += qr[d] * kr[d] + qi[d] * ki[d];
    float gate = sigf(gdot);
    float gl[3];
    for (int j = 0; j < 3; ++j) {
      float s = 0.f;
      for (int k = 0; k < TWOD; ++k) s += p.z_f32[b * TWOD + k] * p.arb_W[j * TWOD + k];
      gl[j] = s + p.arb_b[j];
    }
    float mx = fmaxf(gl[0], fmaxf(gl[1], gl[2]));
    float e0 = expf(gl[0] - mx), e1 = expf(gl[1] - mx), e2 = expf(gl[2] - mx);
    float inv = 1.f / (e0 + e1 + e2);
    float G0 = e0 * inv, G1 = e1 * inv, G2 = e2 * inv;
    float act = p.active[b];
    float dsum = 0.f;
    for (int d = 0; d < D_; ++d) {
      float cr = p.z_f32[b * TWOD + d], ci = p.z_f32[b * TWOD + D_ + d];
      float ur = G0 * (vr[d] * gate) + G1 * p.m_r[b * D_ + d] + G2 * p.zq[b * TWOD + d];
      float ui = G0 * (vi[d] * gate) + G1 * p.m_i[b * D_ + d] + G2 * p.zq[b * TWOD + D_ + d];
      float car = 0.6f * cr + 0.4f * ur;
      float cai = 0.6f * ci + 0.4f * ui;
      float ang = atan2f(cai, car);
      float df = fabsf(ang - p.prev_ang[b * D_ + d]);
      df = fminf(df, TWOPI - df);
      dsum += df;
      p.prev_ang[b * D_ + d] = ang;
      if (act > 0.5f) { p.gw_r[b * D_ + d] = car; p.gw_i[b * D_ + d] = cai; }
    }
    p.acc_ph[b]  += act * dsum * (1.f / D_);
    p.acc_pond[b] += act * 0.01f;
    if (act > 0.5f) { p.acc_vq[b] = p.vql[b]; p.acc_ent[b] = s_ent; p.fin_idx[b] = p.idx[b]; }
    float biasSP = log1pf(expf(p.halt_bias[0]));
    float hp = sigf(biasSP - p.vql[b]);
    p.active[b] = (hp > 0.5f) ? 0.f : act;
  }

  // I: end-of-timestep — memory write, outputs, stats
  if (rec == 7) {
    __syncthreads();
    if (tid < B_) {
      int b = tid;
      float s = 0.f;
      for (int k = 0; k < D_; ++k)
        s += p.gw_r[b * D_ + k] * p.gate_w[k] + p.gw_i[b * D_ + k] * p.gate_w[D_ + k];
      p.vql[b] = sigf(s + p.gate_b[0]);  // reuse vql as write-gate scratch
      for (int d = 0; d < D_; ++d) {
        p.out_bf[((size_t)b * T_ + t) * TWOD + d]       = (__bf16)p.gw_r[b * D_ + d];
        p.out_bf[((size_t)b * T_ + t) * TWOD + D_ + d]  = (__bf16)p.gw_i[b * D_ + d];
      }
      p.out[MV + 4 + (size_t)b * T_ + t] = (float)p.fin_idx[b];
    }
    __syncthreads();
    for (int c = tid; c < B_ * D_; c += 256) {
      int b = c >> 6, d = c & 63;
      float wg = p.vql[b];
      size_t base = (size_t)b * S_ * D_ + d;
      float lr = p.mem_r[base + (S_ - 1) * D_];
      float li = p.mem_i[base + (S_ - 1) * D_];
      for (int s2 = S_ - 1; s2 >= 1; --s2) {
        p.mem_r[base + s2 * D_] = p.mem_r[base + (s2 - 1) * D_];
        p.mem_i[base + s2 * D_] = p.mem_i[base + (s2 - 1) * D_];
      }
      p.mem_r[base] = wg * p.gw_r[b * D_ + d] + (1.f - wg) * lr;
      p.mem_i[base] = wg * p.gw_i[b * D_ + d] + (1.f - wg) * li;
    }
    __syncthreads();
    if (tid < 4) {
      float* accs[4] = {p.acc_vq, p.acc_ent, p.acc_pond, p.acc_ph};
      float s = 0.f;
      for (int b = 0; b < B_; ++b) s += accs[tid][b];
      p.stats[tid] += s * (1.f / B_);
    }
  }
}

// ---- decoder GEMM: (4096 x 128) x (128 x 8192) via bf16 WMMA ----------------
// B tile (64 rows x 128 k) staged once into LDS per WG via async copy.
__global__ void decode_k(P p) {
  __shared__ __align__(16) __bf16 Bt[64 * TWOD]; // 16 KB
  int tid = threadIdx.x;
  int lane = tid & 31, w = tid >> 5;
  int r = lane & 15, half = lane >> 4;
  size_t m0 = (size_t)blockIdx.y * 128, n0 = (size_t)blockIdx.x * 64;

  stage_to_lds((const char*)(p.dec_bf + n0 * TWOD), (char*)Bt,
               (64 * TWOD * 2) / 16, tid);

  const __bf16* arow = p.out_bf + (m0 + 16 * w + r) * TWOD + half * 8;
  v16bf a[4];
#pragma unroll
  for (int kk = 0; kk < 4; ++kk) a[kk] = ld_op(arow + 32 * kk);
  __syncthreads();

  for (int nt = 0; nt < 4; ++nt) {
    size_t n = n0 + nt * 16 + r;
    const __bf16* brow = Bt + (size_t)(nt * 16 + r) * TWOD + half * 8;
    v8f acc = {0.f, 0.f, 0.f, 0.f, 0.f, 0.f, 0.f, 0.f};
#pragma unroll
    for (int kk = 0; kk < 4; ++kk) acc = wmma_bf16(a[kk], ld_op(brow + 32 * kk), acc);
    float bias = p.dec_b[n];
#pragma unroll
    for (int v = 0; v < 8; ++v) {
      size_t m = m0 + 16 * w + v + half * 8;
      p.out[m * V_ + n] = acc[v] + bias;
    }
  }
}

__global__ void stats_out(P p) {
  if (threadIdx.x < 4) p.out[MV + threadIdx.x] = p.stats[threadIdx.x] * (1.f / T_);
}

// ---- host -------------------------------------------------------------------
extern "C" void kernel_launch(void* const* d_in, const int* in_sizes, int n_in,
                              void* d_out, int out_size, void* d_ws, size_t ws_size,
                              hipStream_t stream) {
  (void)in_sizes; (void)n_in; (void)out_size; (void)ws_size;
  P p{};
  p.x_seq = (const int*)d_in[0];
  p.enc   = (const float*)d_in[1];
  p.vq    = (const float*)d_in[2];
  p.nrg   = (const float*)d_in[3];  p.nrb = (const float*)d_in[4];
  p.nig   = (const float*)d_in[5];  p.nib = (const float*)d_in[6];
  p.qkv_Wr = (const float*)d_in[7];  p.qkv_br = (const float*)d_in[8];
  p.qkv_Wi = (const float*)d_in[9];  p.qkv_bi = (const float*)d_in[10];
  p.gate_w = (const float*)d_in[11]; p.gate_b = (const float*)d_in[12];
  p.arb_W  = (const float*)d_in[13]; p.arb_b  = (const float*)d_in[14];
  p.dec_W  = (const float*)d_in[15]; p.dec_b  = (const float*)d_in[16];
  p.halt_bias = (const float*)d_in[17];
  p.input_gate = (const float*)d_in[18];

  char* w = (char*)d_ws;
  auto take = [&](size_t bytes) -> char* {
    char* r = w; w += (bytes + 255) & ~(size_t)255; return r;
  };
  p.x_r   = (float*)take((size_t)T_ * B_ * D_ * 4);
  p.x_i   = (float*)take((size_t)T_ * B_ * D_ * 4);
  p.gw_r  = (float*)take(B_ * D_ * 4);
  p.gw_i  = (float*)take(B_ * D_ * 4);
  p.mem_r = (float*)take((size_t)B_ * S_ * D_ * 4);
  p.mem_i = (float*)take((size_t)B_ * S_ * D_ * 4);
  p.z_f32 = (float*)take(B_ * TWOD * 4);
  p.prev_ang = (float*)take(B_ * D_ * 4);
  p.active  = (float*)take(B_ * 4);
  p.acc_vq  = (float*)take(B_ * 4);
  p.acc_ent = (float*)take(B_ * 4);
  p.acc_pond = (float*)take(B_ * 4);
  p.acc_ph  = (float*)take(B_ * 4);
  p.fin_idx = (int*)take(B_ * 4);
  p.vql  = (float*)take(B_ * 4);
  p.sim  = (float*)take(B_ * S_ * 4);
  p.m_r  = (float*)take(B_ * D_ * 4);
  p.m_i  = (float*)take(B_ * D_ * 4);
  p.qkv  = (float*)take(6 * B_ * D_ * 4);
  p.e_sq = (float*)take(V_ * 4);
  p.part_val = (float*)take(64 * B_ * 4);
  p.part_idx = (int*)take(64 * B_ * 4);
  p.idx    = (int*)take(B_ * 4);
  p.counts = (int*)take(V_ * 4);
  p.zq     = (float*)take(B_ * TWOD * 4);
  p.stats  = (float*)take(4 * 4);
  p.b_re   = (float*)take(3 * D_ * 4);
  p.b_im   = (float*)take(3 * D_ * 4);
  p.z_bf   = (__bf16*)take((size_t)B_ * TWOD * 2);
  p.vq_bf  = (__bf16*)take((size_t)V_ * TWOD * 2);
  p.dec_bf = (__bf16*)take((size_t)V_ * TWOD * 2);
  p.out_bf = (__bf16*)take((size_t)B_ * T_ * TWOD * 2);
  p.Wre    = (__bf16*)take((size_t)3 * D_ * TWOD * 2);
  p.Wim    = (__bf16*)take((size_t)3 * D_ * TWOD * 2);
  p.out = (float*)d_out;

  prep_zero<<<256, 256, 0, stream>>>(p);
  prep_cvt<<<1024, 256, 0, stream>>>(p);
  prep_esq<<<32, 256, 0, stream>>>(p);
  prep_packw<<<96, 256, 0, stream>>>(p);
  embed_k<<<16, 256, 0, stream>>>(p);

  for (int t = 0; t < T_; ++t) {
    for (int r = 0; r < 8; ++r) {
      step_pre<<<1, 128, 0, stream>>>(p, t, r);
      vq_search<<<64, 256, 0, stream>>>(p);
      step_main<<<1, 256, 0, stream>>>(p, t, r);
    }
  }

  decode_k<<<dim3(V_ / 64, (B_ * T_) / 128), 256, 0, stream>>>(p);
  stats_out<<<1, 32, 0, stream>>>(p);
}
